// LlamaQDecoderLayer_54365696033095
// MI455X (gfx1250) — matrix operand order repaired and checked by reading, not compiled, not run on hardware
//
#include <hip/hip_runtime.h>
#include <hip/hip_bf16.h>

// ---------------------------------------------------------------------------
// Llama AWQ decoder layer for MI455X (gfx1250, wave32, WMMA).
// T=1024, H=4096, NH=32, HD=128, I=11008, AWQ group G=128.
// ~415 GFLOP of GEMM -> f16 WMMA with on-the-fly int4 dequant into LDS.
// Block tile 128x128, K-step 64; wave tile 32x64 = 2x4 WMMA 16x16x32 tiles.
// A-tile staged with GLOBAL_LOAD_ASYNC_TO_LDS_B128 (ASYNCcnt path).
// ---------------------------------------------------------------------------

typedef __attribute__((ext_vector_type(16))) _Float16 v16h;
typedef __attribute__((ext_vector_type(8)))  float    v8f;

#define T_TOK 1024
#define H_DIM 4096
#define NH    32
#define HD    128
#define I_DIM 11008
#define QKV_N (3 * H_DIM)
#define EPS   1e-6f
#define LN10000 9.210340371976184f  // ln(10000)

#define LDS_PITCH 72               // halves; 144B rows (16B mult, bank-cycling)
#define A_TILE_HALVES (128 * LDS_PITCH)

// ---------------------------------------------------------------------------
// RMSNorm: f32 in -> f16 out (feeds WMMA A-matrix). One block per row.
// ---------------------------------------------------------------------------
__global__ __launch_bounds__(256) void rmsnorm_f16_kernel(
    const float* __restrict__ x, const float* __restrict__ w,
    _Float16* __restrict__ o) {
  __shared__ float red[8];
  const int row = blockIdx.x;
  const int tid = threadIdx.x;
  const float* xr = x + (size_t)row * H_DIM;
  float ss = 0.f;
  for (int i = tid; i < H_DIM; i += 256) {
    float v = xr[i];
    ss += v * v;
  }
  #pragma unroll
  for (int off = 16; off > 0; off >>= 1) ss += __shfl_xor(ss, off, 32);
  if ((tid & 31) == 0) red[tid >> 5] = ss;
  __syncthreads();
  if (tid == 0) {
    float t = 0.f;
    #pragma unroll
    for (int j = 0; j < 8; ++j) t += red[j];
    red[0] = rsqrtf(t * (1.0f / H_DIM) + EPS);
  }
  __syncthreads();
  const float rn = red[0];
  _Float16* orow = o + (size_t)row * H_DIM;
  for (int i = tid; i < H_DIM; i += 256)
    orow[i] = (_Float16)(xr[i] * rn * w[i]);
}

// ---------------------------------------------------------------------------
// AWQ int4 dequant + WMMA f16 GEMM.  C[M,N] = A[M,K] @ deq(W)[K,N] (+res).
// Block: 256 threads = 8 waves. Block tile 128(M) x 128(N), K-step 64.
// Wave tile 32x64 = 2x4 WMMA tiles -> 16 WMMA per barrier pair.
// A tile: async global->LDS DMA (no VGPR round trip, ASYNCcnt).
// B tile: dequant in-register, packed b64 stores along k; stored [n][k] so
// fragments are contiguous b128 LDS reads (ISA 7.12.2 layouts).
// ---------------------------------------------------------------------------
__global__ __launch_bounds__(256) void gemm_awq_wmma_kernel(
    const _Float16* __restrict__ A,
    const int* __restrict__ qw,      // [K, N/8] packed nibbles
    const int* __restrict__ qz,      // [K/128, N/8]
    const float* __restrict__ sc,    // [K/128, N]
    const float* __restrict__ residual,  // [M, N] or nullptr
    float* __restrict__ C,           // [M, N]
    int M, int N, int K) {
  // Single LDS block so async-copy byte offsets are deterministic (seg base 0).
  __shared__ __align__(16) _Float16 smem[2 * A_TILE_HALVES];
  _Float16 (*As)[LDS_PITCH] = reinterpret_cast<_Float16(*)[LDS_PITCH]>(smem);
  _Float16 (*Bs)[LDS_PITCH] =
      reinterpret_cast<_Float16(*)[LDS_PITCH]>(smem + A_TILE_HALVES);

  const int tid  = threadIdx.x;
  const int lane = tid & 31;
  const int wave = tid >> 5;
  const int wm = wave & 3;           // 4 waves along M (32 rows each)
  const int wn = wave >> 2;          // 2 waves along N (64 cols each)
  const int g16 = lane >> 4;         // half-wave select
  const int l16 = lane & 15;

  const int m0 = blockIdx.y * 128;
  const int n0 = blockIdx.x * 128;
  const int n8 = N >> 3;

  // cooperative-load roles
  const int arow = tid >> 1;             // 0..127
  const int aseg = (tid & 1) * 32;       // 0 or 32 halves
  const int bj   = tid & 15;             // packed-dword column (8 n each)
  const int bk4  = (tid >> 4) * 4;       // k base within tile: 0,4,..,60

  // LDS byte offset of this thread's A-tile destination (segment-relative).
  const unsigned lds_a = (unsigned)(arow * LDS_PITCH + aseg) * 2u;

  v8f acc[2][4] = {};

  for (int k0 = 0; k0 < K; k0 += 64) {
    // ---- stage A tile via async DMA: 4x b128, offset hits LDS+global ----
    {
      const _Float16* gsrc = A + (size_t)(m0 + arow) * K + k0 + aseg;
      asm volatile(
          "global_load_async_to_lds_b128 %0, %1, off\n\t"
          "global_load_async_to_lds_b128 %0, %1, off offset:16\n\t"
          "global_load_async_to_lds_b128 %0, %1, off offset:32\n\t"
          "global_load_async_to_lds_b128 %0, %1, off offset:48"
          :
          : "v"(lds_a), "v"(gsrc)
          : "memory");
    }
    // ---- dequant B tile: 64 k-rows x 128 n-cols ----
    // Thread owns (8 n-cols) x (4 k-rows): 4 packed dwords in, 8 b64 out.
    {
      const int grp = k0 >> 7;  // 64-deep slice never straddles a 128-group
      const size_t qb = (size_t)(k0 + bk4) * n8 + (n0 >> 3) + bj;
      const int pw0 = qw[qb];
      const int pw1 = qw[qb + n8];
      const int pw2 = qw[qb + 2 * (size_t)n8];
      const int pw3 = qw[qb + 3 * (size_t)n8];
      const int pz  = qz[(size_t)grp * n8 + (n0 >> 3) + bj];
      const float* scp = sc + (size_t)grp * N + n0 + bj * 8;
      if (k0 + 64 < K)
        __builtin_prefetch(qw + qb + 64 * (size_t)n8, 0, 1);
      #pragma unroll
      for (int i = 0; i < 8; ++i) {
        const int zv = (pz >> (4 * i)) & 0xF;
        const float s = scp[i];
        union { _Float16 h[4]; uint2 u; } pk;
        pk.h[0] = (_Float16)((float)(((pw0 >> (4 * i)) & 0xF) - zv) * s);
        pk.h[1] = (_Float16)((float)(((pw1 >> (4 * i)) & 0xF) - zv) * s);
        pk.h[2] = (_Float16)((float)(((pw2 >> (4 * i)) & 0xF) - zv) * s);
        pk.h[3] = (_Float16)((float)(((pw3 >> (4 * i)) & 0xF) - zv) * s);
        *reinterpret_cast<uint2*>(&Bs[bj * 8 + i][bk4]) = pk.u;
      }
    }
    // drain this wave's async copies, then rendezvous
    asm volatile("s_wait_asynccnt 0" ::: "memory");
    __syncthreads();

    // ---- two WMMA K-steps of 32 ----
    union Frag { v16h v; uint4 u[2]; };
    #pragma unroll
    for (int ks = 0; ks < 64; ks += 32) {
      Frag a[2], b[4];
      #pragma unroll
      for (int s = 0; s < 2; ++s) {
        const int m = 32 * wm + 16 * s + l16;
        // A frag: halves 0-7 <- K=ks+8*g16.., halves 8-15 <- K=ks+16+8*g16..
        a[s].u[0] = *reinterpret_cast<const uint4*>(&As[m][ks + 8 * g16]);
        a[s].u[1] = *reinterpret_cast<const uint4*>(&As[m][ks + 16 + 8 * g16]);
      }
      #pragma unroll
      for (int t = 0; t < 4; ++t) {
        const int n = 64 * wn + 16 * t + l16;
        // B frag: halves 0-7 <- K=ks+8*g16.., halves 8-15 <- K=ks+16+8*g16..
        b[t].u[0] = *reinterpret_cast<const uint4*>(&Bs[n][ks + 8 * g16]);
        b[t].u[1] = *reinterpret_cast<const uint4*>(&Bs[n][ks + 16 + 8 * g16]);
      }
      #pragma unroll
      for (int s = 0; s < 2; ++s)
        #pragma unroll
        for (int t = 0; t < 4; ++t)
          acc[s][t] = __builtin_amdgcn_wmma_f32_16x16x32_f16(
              false, a[s].v, false, b[t].v, (short)0, acc[s][t], false, false);
    }
    __syncthreads();
  }

  // ---- epilogue: C layout lane=N (l16), VGPR r -> M = r + 8*g16 ----
  #pragma unroll
  for (int s = 0; s < 2; ++s) {
    const int mbase = m0 + 32 * wm + 16 * s + 8 * g16;
    #pragma unroll
    for (int t = 0; t < 4; ++t) {
      const int col = n0 + 64 * wn + 16 * t + l16;
      #pragma unroll
      for (int r = 0; r < 8; ++r) {
        size_t idx = (size_t)(mbase + r) * N + col;
        float v = acc[s][t][r];
        if (residual) v += residual[idx];
        C[idx] = v;
      }
    }
  }
}

// ---------------------------------------------------------------------------
// RoPE (neox half-rotation) on q,k + f32->f16 split of q/k/v.
// One thread per (t, h, i) with i in [0, HD/2).
// ---------------------------------------------------------------------------
__global__ __launch_bounds__(256) void rope_split_kernel(
    const float* __restrict__ qkv, const int* __restrict__ pos,
    _Float16* __restrict__ qh, _Float16* __restrict__ kh,
    _Float16* __restrict__ vh) {
  const int idx = blockIdx.x * 256 + threadIdx.x;  // T*NH*64
  const int i = idx & 63;
  const int h = (idx >> 6) & (NH - 1);
  const int t = idx >> 11;
  const float p = (float)pos[t];
  const float freq = p * __expf(-(float)(2 * i) * (LN10000 / (float)HD));
  float s, c;
  __sincosf(freq, &s, &c);
  const size_t base = (size_t)t * QKV_N + (size_t)h * HD;
  const float q1 = qkv[base + i],          q2 = qkv[base + 64 + i];
  const float k1 = qkv[base + H_DIM + i],  k2 = qkv[base + H_DIM + 64 + i];
  const size_t ob = (size_t)t * H_DIM + (size_t)h * HD;
  qh[ob + i]      = (_Float16)(q1 * c - q2 * s);
  qh[ob + 64 + i] = (_Float16)(q2 * c + q1 * s);
  kh[ob + i]      = (_Float16)(k1 * c - k2 * s);
  kh[ob + 64 + i] = (_Float16)(k2 * c + k1 * s);
  vh[ob + i]      = (_Float16)qkv[base + 2 * H_DIM + i];
  vh[ob + 64 + i] = (_Float16)qkv[base + 2 * H_DIM + 64 + i];
}

// ---------------------------------------------------------------------------
// Causal attention: one wave32 per (head, query), online softmax.
// Lane owns 4 of 128 head dims via b64 vector loads; score via butterfly
// shfl_xor reduction. K/V (16 MB for the layer) live in L2.
// ---------------------------------------------------------------------------
__global__ __launch_bounds__(256) void attn_kernel(
    const _Float16* __restrict__ Q, const _Float16* __restrict__ K,
    const _Float16* __restrict__ V, _Float16* __restrict__ O) {
  const int wid  = blockIdx.x * 8 + (threadIdx.x >> 5);
  const int lane = threadIdx.x & 31;
  const int q = wid & (T_TOK - 1);
  const int h = wid >> 10;
  const size_t hb = (size_t)h * HD + lane * 4;

  union H4 { uint2 u; _Float16 h[4]; };
  H4 qv;
  qv.u = *reinterpret_cast<const uint2*>(Q + (size_t)q * H_DIM + hb);
  const float scale = 0.08838834764831845f;  // HD^-0.5
  const float qv0 = (float)qv.h[0] * scale, qv1 = (float)qv.h[1] * scale;
  const float qv2 = (float)qv.h[2] * scale, qv3 = (float)qv.h[3] * scale;

  float a0 = 0.f, a1 = 0.f, a2 = 0.f, a3 = 0.f;
  float m = -INFINITY, l = 0.f;
  for (int t = 0; t <= q; ++t) {
    H4 kv, vv;
    kv.u = *reinterpret_cast<const uint2*>(K + (size_t)t * H_DIM + hb);
    float s = qv0 * (float)kv.h[0] + qv1 * (float)kv.h[1] +
              qv2 * (float)kv.h[2] + qv3 * (float)kv.h[3];
    #pragma unroll
    for (int off = 16; off > 0; off >>= 1) s += __shfl_xor(s, off, 32);
    const float mn = fmaxf(m, s);
    const float ao = __expf(m - mn);
    const float p  = __expf(s - mn);
    vv.u = *reinterpret_cast<const uint2*>(V + (size_t)t * H_DIM + hb);
    a0 = a0 * ao + p * (float)vv.h[0];
    a1 = a1 * ao + p * (float)vv.h[1];
    a2 = a2 * ao + p * (float)vv.h[2];
    a3 = a3 * ao + p * (float)vv.h[3];
    l = l * ao + p;
    m = mn;
  }
  const float inv = 1.0f / l;
  H4 ov;
  ov.h[0] = (_Float16)(a0 * inv);
  ov.h[1] = (_Float16)(a1 * inv);
  ov.h[2] = (_Float16)(a2 * inv);
  ov.h[3] = (_Float16)(a3 * inv);
  *reinterpret_cast<uint2*>(O + (size_t)q * H_DIM + hb) = ov.u;
}

// ---------------------------------------------------------------------------
// act = silu(gate) * up, f32 -> f16 (feeds down-proj WMMA A-matrix)
// ---------------------------------------------------------------------------
__global__ __launch_bounds__(256) void silu_mul_kernel(
    const float* __restrict__ g, const float* __restrict__ u,
    _Float16* __restrict__ o) {
  const int i = blockIdx.x * 256 + threadIdx.x;
  const float gv = g[i];
  const float uv = u[i];
  o[i] = (_Float16)((gv / (1.0f + __expf(-gv))) * uv);
}

// ---------------------------------------------------------------------------
// Launch: full layer pipeline on `stream`. Workspace layout (reused):
//   [0, 50.3M)      qkv f32            -> reused by gate f32 (45.1M)
//   [50.3M, 75.5M)  q/k/v f16 (3x8M)   -> reused by act f16 (22.5M)
//   [75.5M, 120.6M) up f32
//   [120.6M,129.0M) xn f16 (ln1, reused for ln2)
//   [129.0M,137.4M) attn f16
//   [137.4M,154.1M) hidden f32 (post-attention residual)
// ---------------------------------------------------------------------------
extern "C" void kernel_launch(void* const* d_in, const int* in_sizes, int n_in,
                              void* d_out, int out_size, void* d_ws, size_t ws_size,
                              hipStream_t stream) {
  const int*   positions = (const int*)d_in[0];
  const float* hidden_in = (const float*)d_in[1];
  const float* ln1_w     = (const float*)d_in[2];
  const float* ln2_w     = (const float*)d_in[3];
  const int*   qkv_qw = (const int*)d_in[4];
  const int*   qkv_qz = (const int*)d_in[5];
  const float* qkv_sc = (const float*)d_in[6];
  const int*   o_qw   = (const int*)d_in[7];
  const int*   o_qz   = (const int*)d_in[8];
  const float* o_sc   = (const float*)d_in[9];
  const int*   g_qw   = (const int*)d_in[10];
  const int*   g_qz   = (const int*)d_in[11];
  const float* g_sc   = (const float*)d_in[12];
  const int*   u_qw   = (const int*)d_in[13];
  const int*   u_qz   = (const int*)d_in[14];
  const float* u_sc   = (const float*)d_in[15];
  const int*   d_qw   = (const int*)d_in[16];
  const int*   d_qz   = (const int*)d_in[17];
  const float* d_sc   = (const float*)d_in[18];
  float* out = (float*)d_out;

  char* ws = (char*)d_ws;
  const size_t OFF_QKV = 0;                                    // 50,331,648 B
  const size_t OFF_QF  = OFF_QKV + (size_t)T_TOK * QKV_N * 4;  // q f16
  const size_t OFF_KF  = OFF_QF + (size_t)T_TOK * H_DIM * 2;
  const size_t OFF_VF  = OFF_KF + (size_t)T_TOK * H_DIM * 2;
  const size_t OFF_UP  = OFF_VF + (size_t)T_TOK * H_DIM * 2;   // up f32
  const size_t OFF_XN  = OFF_UP + (size_t)T_TOK * I_DIM * 4;   // xn f16
  const size_t OFF_ATT = OFF_XN + (size_t)T_TOK * H_DIM * 2;   // attn f16
  const size_t OFF_HID = OFF_ATT + (size_t)T_TOK * H_DIM * 2;  // hidden f32
  const size_t OFF_GATE = OFF_QKV;  // reuse qkv f32 region
  const size_t OFF_ACT  = OFF_QF;   // reuse q/k/v f16 region

  float*     qkv_f32  = (float*)(ws + OFF_QKV);
  _Float16*  q_f16    = (_Float16*)(ws + OFF_QF);
  _Float16*  k_f16    = (_Float16*)(ws + OFF_KF);
  _Float16*  v_f16    = (_Float16*)(ws + OFF_VF);
  float*     up_f32   = (float*)(ws + OFF_UP);
  _Float16*  xn_f16   = (_Float16*)(ws + OFF_XN);
  _Float16*  attn_f16 = (_Float16*)(ws + OFF_ATT);
  float*     hid_f32  = (float*)(ws + OFF_HID);
  float*     gate_f32 = (float*)(ws + OFF_GATE);
  _Float16*  act_f16  = (_Float16*)(ws + OFF_ACT);

  // 1) RMSNorm(hidden, ln1) -> f16
  rmsnorm_f16_kernel<<<T_TOK, 256, 0, stream>>>(hidden_in, ln1_w, xn_f16);

  // 2) QKV projection: [1024,4096] x [4096,12288]
  gemm_awq_wmma_kernel<<<dim3(QKV_N / 128, T_TOK / 128), 256, 0, stream>>>(
      xn_f16, qkv_qw, qkv_qz, qkv_sc, nullptr, qkv_f32, T_TOK, QKV_N, H_DIM);

  // 3) RoPE + split into q/k/v f16
  rope_split_kernel<<<(T_TOK * NH * (HD / 2)) / 256, 256, 0, stream>>>(
      qkv_f32, positions, q_f16, k_f16, v_f16);

  // 4) Causal attention (one wave per head x query)
  attn_kernel<<<(NH * T_TOK) / 8, 256, 0, stream>>>(q_f16, k_f16, v_f16,
                                                    attn_f16);

  // 5) O projection + residual(hidden_in) -> hid_f32
  gemm_awq_wmma_kernel<<<dim3(H_DIM / 128, T_TOK / 128), 256, 0, stream>>>(
      attn_f16, o_qw, o_qz, o_sc, hidden_in, hid_f32, T_TOK, H_DIM, H_DIM);

  // 6) RMSNorm(hid, ln2) -> f16
  rmsnorm_f16_kernel<<<T_TOK, 256, 0, stream>>>(hid_f32, ln2_w, xn_f16);

  // 7) gate / up projections: [1024,4096] x [4096,11008]
  gemm_awq_wmma_kernel<<<dim3(I_DIM / 128, T_TOK / 128), 256, 0, stream>>>(
      xn_f16, g_qw, g_qz, g_sc, nullptr, gate_f32, T_TOK, I_DIM, H_DIM);
  gemm_awq_wmma_kernel<<<dim3(I_DIM / 128, T_TOK / 128), 256, 0, stream>>>(
      xn_f16, u_qw, u_qz, u_sc, nullptr, up_f32, T_TOK, I_DIM, H_DIM);

  // 8) silu(gate)*up -> f16
  silu_mul_kernel<<<(T_TOK * I_DIM) / 256, 256, 0, stream>>>(gate_f32, up_f32,
                                                             act_f16);

  // 9) down projection + residual(hid) -> d_out
  gemm_awq_wmma_kernel<<<dim3(H_DIM / 128, T_TOK / 128), 256, 0, stream>>>(
      act_f16, d_qw, d_qz, d_sc, hid_f32, out, T_TOK, H_DIM, I_DIM);

  (void)in_sizes; (void)n_in; (void)out_size; (void)ws_size;
}
